// AttentionOp_41042707480859
// MI455X (gfx1250) — matrix operand order repaired and checked by reading, not compiled, not run on hardware
//
#include <hip/hip_runtime.h>
#include <cstddef>
#include <cstdint>

// Sliding-window attention (window=256) with pre/post-softmax talking-heads
// cross-head projection, for gfx1250 (MI455X).
//  - QK^T and P*V: v_wmma_f32_16x16x32_bf16 (f32 accumulate)
//  - static 16x16 head-mixing: v_wmma_f32_16x16x4_f32 over each (q,s) tile
//  - rank-2 dynamic terms, diagonal gating, softmax: f32 VALU
//
// Shapes (fixed by the reference): B=2, T=S=1024, N=16 heads, D=128,
// G=1, M=16, I=2, WINDOW=256.
//
// Out-of-range global addresses are CLAMPED, never guarded: clamped K rows
// only produce logits that are overwritten by the mask, and clamped V rows
// are only multiplied by probabilities that are exactly zero.

#define T_SEQ   1024
#define NHEADS  16
#define DHEAD   128
#define WINDOW  256
#define NEG_BIG (-3.0e38f)

typedef __bf16 bf16;
typedef bf16  v16bf __attribute__((ext_vector_type(16)));
typedef bf16  v8bf  __attribute__((ext_vector_type(8)));
typedef float v8f   __attribute__((ext_vector_type(8)));
typedef float v4f   __attribute__((ext_vector_type(4)));
typedef float v2f   __attribute__((ext_vector_type(2)));

// Dynamic part of the cross-head projection at one (t,s) position, producing
// heads NB..NB+7:  y = x + mixed + rank-2 dynamic + diagonal gating.
// (The static x*W term arrives pre-computed in mrow via f32 WMMA.)
template <int NB>
__device__ __forceinline__ void chdyn8(const float x[16],
                                       const float* __restrict__ mrow, // mx[qq][ss][*]
                                       const float* __restrict__ qw1,
                                       const float* __restrict__ qw2,
                                       const float* __restrict__ kw1,
                                       const float* __restrict__ kw2,
                                       const float* __restrict__ qdd,
                                       const float* __restrict__ kdd,
                                       float y[8])
{
  float a0 = 0.f, a1 = 0.f, b0 = 0.f, b1 = 0.f;
#pragma unroll
  for (int m = 0; m < 16; ++m) {
    a0 += x[m] * qw1[2*m  ];
    a1 += x[m] * qw1[2*m+1];
    b0 += x[m] * kw1[2*m  ];
    b1 += x[m] * kw1[2*m+1];
  }
#pragma unroll
  for (int j = 0; j < 8; ++j) {
    const int nn = NB + j;
    float acc = x[nn] + mrow[nn];
    acc += a0 * qw2[2*nn] + a1 * qw2[2*nn+1];
    acc += b0 * kw2[2*nn] + b1 * kw2[2*nn+1];
    acc += x[nn] * (qdd[nn] + kdd[nn]);
    y[j] = acc;
  }
}

__global__ __launch_bounds__(512, 1)
void slide_talking_heads_attn(const float* __restrict__ q,
                              const float* __restrict__ k,
                              const float* __restrict__ v,
                              const float* __restrict__ w_pre,
                              const float* __restrict__ w_post,
                              const float* __restrict__ qw1_pre,
                              const float* __restrict__ qw2_pre,
                              const float* __restrict__ kw1_pre,
                              const float* __restrict__ kw2_pre,
                              const float* __restrict__ qw1_post,
                              const float* __restrict__ qw2_post,
                              const float* __restrict__ kw1_post,
                              const float* __restrict__ kw2_post,
                              const float* __restrict__ qdd_pre,
                              const float* __restrict__ kdd_pre,
                              const float* __restrict__ qdd_post,
                              const float* __restrict__ kdd_post,
                              float* __restrict__ out)
{
  __shared__ float lt[16][16][17];             // (head, q, s) tile, padded
  __shared__ __align__(16) float mx[16][16][16]; // (q, s, head) mixed tile
  __shared__ float rowm[16][16];               // running max per (head, q)
  __shared__ float rowz[16][16];               // sum-exp, then its reciprocal
  __shared__ __align__(16) bf16 pp[16][16][32];// projected probs (pair tiles)
  __shared__ float wpreS[256];
  __shared__ float wpostS[256];

  const int tid  = threadIdx.x;
  const int wave = tid >> 5;                   // one wave per head for WMMA
  const int lane = tid & 31;
  const int r    = lane & 15;
  const int h    = lane >> 4;
  const int b    = blockIdx.x >> 6;            // grid = B * (T/16) = 128
  const int t0   = (blockIdx.x & 63) << 4;
  const int n    = wave;
  const bool hiHalf = (wave >= 8);             // wave-uniform head split

  // projection-phase coordinates: two waves cooperate on each (q,s) position
  const int qq = (tid >> 4) & 15;
  const int ss = tid & 15;
  const int tq = t0 + qq;

  for (int i = tid; i < 256; i += 512) { wpreS[i] = w_pre[i]; wpostS[i] = w_post[i]; }
  if (tid < 256) { rowm[tid >> 4][tid & 15] = NEG_BIG; rowz[tid >> 4][tid & 15] = 0.f; }
  __syncthreads();

  // t-dependent dynamic-weight rows (fixed per thread): hoisted
  const size_t trow = (size_t)(b * T_SEQ + tq);
  const float* q1pre  = qw1_pre  + trow * 32;
  const float* q2pre  = qw2_pre  + trow * 32;
  const float* qdpre  = qdd_pre  + trow * 16;
  const float* q1post = qw1_post + trow * 32;
  const float* q2post = qw2_post + trow * 32;
  const float* qdpost = qdd_post + trow * 16;

  const float invsq = 0.08838834764831845f;    // 1/sqrt(128)

  // QK^T for one 16-key tile starting at s0.  Q re-loaded each call (L0-hot)
  // to keep register pressure low.  All loads unconditional float4 (b128).
  auto qk_tile = [&](int s0) -> v8f {
    v8f acc;
#pragma unroll
    for (int i = 0; i < 8; ++i) acc[i] = 0.f;
    int krow = s0 + r;
    if (krow < 0) krow = 0;                    // clamped; masked later anyway
    const v4f* kb = (const v4f*)(k + (((size_t)(b * T_SEQ + krow)) * NHEADS + n) * DHEAD) + 2 * h;
    const v4f* qb = (const v4f*)(q + (((size_t)(b * T_SEQ + t0 + r)) * NHEADS + n) * DHEAD) + 2 * h;
#pragma unroll
    for (int c = 0; c < 4; ++c) {
      // A (dense 16-bit layout): elems 0..7 -> K=32c+8h+e, 8..15 -> +16
      v4f a0 = qb[8*c + 0], a1 = qb[8*c + 1], a2 = qb[8*c + 4], a3 = qb[8*c + 5];
      // B: lane holds key column, elems 0..7 -> K=32c+8h+e, 8..15 -> +16
      v4f k0 = kb[8*c + 0], k1 = kb[8*c + 1], k2 = kb[8*c + 4], k3 = kb[8*c + 5];
      v16bf af, bfv;
#pragma unroll
      for (int e = 0; e < 4; ++e) {
        af[e]      = (bf16)a0[e];
        af[4 + e]  = (bf16)a1[e];
        af[8 + e]  = (bf16)a2[e];
        af[12 + e] = (bf16)a3[e];
        bfv[e]      = (bf16)k0[e];
        bfv[4 + e]  = (bf16)k1[e];
        bfv[8 + e]  = (bf16)k2[e];
        bfv[12 + e] = (bf16)k3[e];
      }
      acc = __builtin_amdgcn_wmma_f32_16x16x32_bf16(false, af, false, bfv,
                                                    (short)0, acc, false, false);
    }
#pragma unroll
    for (int i = 0; i < 8; ++i) acc[i] *= invsq;
    return acc;
  };

  // Static head-mixing via f32 WMMA: wave w computes, for query row qq=w,
  // Y[ss][n] = sum_m lt[m][w][ss] * W[m][n] and writes it to mx[w][ss][n].
  // A 16x4 f32 layout: a[0]=X[M=r][K=4c+2h], a[1]=X[M=r][K=4c+2h+1].
  auto mix_tile = [&](const float* __restrict__ wMat) {
    v8f yc;
#pragma unroll
    for (int i = 0; i < 8; ++i) yc[i] = 0.f;
#pragma unroll
    for (int c = 0; c < 4; ++c) {
      v2f a, bw;
      a[0]  = lt[4*c + 2*h    ][wave][r];
      a[1]  = lt[4*c + 2*h + 1][wave][r];
      bw[0] = wMat[(4*c + 2*h    ) * 16 + r];
      bw[1] = wMat[(4*c + 2*h + 1) * 16 + r];
      yc = __builtin_amdgcn_wmma_f32_16x16x4_f32(false, a, false, bw,
                                                 (short)0, yc, false, false);
    }
#pragma unroll
    for (int rr = 0; rr < 8; ++rr) mx[wave][rr + 8 * h][r] = yc[rr];
  };

  // dynamic projection terms at (qq,ss): yv gets this wave-half's 8 heads.
  auto proj_dyn = [&](int s0, const float* kw1b, const float* kw2b, const float* kddb,
                      const float* q1, const float* q2, const float* qd,
                      float yv[8], bool& valid) {
    const int s  = s0 + ss;
    const int sc = s < 0 ? 0 : s;
    valid = (s >= 0) && (s <= tq) && (s > tq - WINDOW);
    const size_t srow = (size_t)(b * T_SEQ + sc);
    float xv[16];
#pragma unroll
    for (int m = 0; m < 16; ++m) xv[m] = lt[m][qq][ss];
    const float* mrow = &mx[qq][ss][0];
    if (hiHalf)
      chdyn8<8>(xv, mrow, q1, q2, kw1b + srow * 32, kw2b + srow * 32,
                qd, kddb + srow * 16, yv);
    else
      chdyn8<0>(xv, mrow, q1, q2, kw1b + srow * 32, kw2b + srow * 32,
                qd, kddb + srow * 16, yv);
  };

  auto write_y = [&](const float yv[8], bool valid) {
    if (hiHalf) {
#pragma unroll
      for (int j = 0; j < 8; ++j) lt[8 + j][qq][ss] = valid ? yv[j] : NEG_BIG;
    } else {
#pragma unroll
      for (int j = 0; j < 8; ++j) lt[j][qq][ss] = valid ? yv[j] : NEG_BIG;
    }
  };

  // =========================== PASS A ===============================
  for (int kt = 0; kt <= 16; ++kt) {
    const int s0 = t0 - WINDOW + 16 * kt;
    if (s0 + 15 < 0) continue;                 // block-uniform

    { // prefetch next key tile (global_prefetch_b8), clamped, unconditional
      int nrow = s0 + 16 + r;
      nrow = nrow < 0 ? 0 : (nrow > T_SEQ - 1 ? T_SEQ - 1 : nrow);
      __builtin_prefetch(k + (((size_t)(b * T_SEQ + nrow)) * NHEADS + n) * DHEAD + 16 * h, 0, 1);
    }

    v8f lg = qk_tile(s0);
#pragma unroll
    for (int rr = 0; rr < 8; ++rr) lt[n][rr + 8 * h][r] = lg[rr];
    __syncthreads();

    mix_tile(wpreS);                           // x*W_pre -> mx (f32 WMMA)
    __syncthreads();

    float yv[8];
    bool  valid;
    proj_dyn(s0, kw1_pre, kw2_pre, kdd_pre, q1pre, q2pre, qdpre, yv, valid);
    __syncthreads();
    write_y(yv, valid);
    __syncthreads();

    if (tid < 256) {                           // one thread per (head, q) row
      const int nn = tid >> 4, qr = tid & 15;
      const float mo = rowm[nn][qr];
      float mxv = mo;
#pragma unroll
      for (int sv = 0; sv < 16; ++sv) mxv = fmaxf(mxv, lt[nn][qr][sv]);
      if (mxv > -1.0e38f) {
        float z = rowz[nn][qr] * __expf(mo - mxv);
#pragma unroll
        for (int sv = 0; sv < 16; ++sv) z += __expf(lt[nn][qr][sv] - mxv);
        rowm[nn][qr] = mxv;
        rowz[nn][qr] = z;
      }
    }
    __syncthreads();
  }

  // convert sum-exp to reciprocal once (kills per-position divisions)
  if (tid < 256) rowz[tid >> 4][tid & 15] = 1.0f / rowz[tid >> 4][tid & 15];
  __syncthreads();

  // =========================== PASS B ===============================
  v8f oacc[8];
#pragma unroll
  for (int ft = 0; ft < 8; ++ft)
#pragma unroll
    for (int i = 0; i < 8; ++i) oacc[ft][i] = 0.f;

  for (int pt = 0; pt <= 8; ++pt) {
    const int s0p = t0 - WINDOW + 32 * pt;
    if (s0p + 31 < 0) continue;                // block-uniform

    for (int hf = 0; hf < 2; ++hf) {
      __syncthreads();                         // pp safe vs previous PV reads
      const int  kt   = 2 * pt + hf;
      const int  s0   = s0p + 16 * hf;
      const bool live = (kt <= 16) && (s0 + 15 >= 0);   // block-uniform
      if (live) {
        v8f lg = qk_tile(s0);
#pragma unroll
        for (int rr = 0; rr < 8; ++rr) lt[n][rr + 8 * h][r] = lg[rr];
        __syncthreads();

        mix_tile(wpreS);                       // x*W_pre -> mx
        __syncthreads();

        float yv[8];
        bool  valid;
        proj_dyn(s0, kw1_pre, kw2_pre, kdd_pre, q1pre, q2pre, qdpre, yv, valid);
        __syncthreads();
        write_y(yv, valid);
        __syncthreads();

        // probabilities in place (each thread owns its 8 head slots here)
        if (hiHalf) {
#pragma unroll
          for (int j = 0; j < 8; ++j) {
            float e = __expf(lt[8 + j][qq][ss] - rowm[8 + j][qq]) * rowz[8 + j][qq];
            lt[8 + j][qq][ss] = valid ? e : 0.f;
          }
        } else {
#pragma unroll
          for (int j = 0; j < 8; ++j) {
            float e = __expf(lt[j][qq][ss] - rowm[j][qq]) * rowz[j][qq];
            lt[j][qq][ss] = valid ? e : 0.f;
          }
        }
        __syncthreads();

        mix_tile(wpostS);                      // p*W_post -> mx
        __syncthreads();

        // dynamic post terms -> projected probs (bf16) into pp
        {
          float py[8];
          bool  pvalid;                        // recomputed; same predicate
          proj_dyn(s0, kw1_post, kw2_post, kdd_post, q1post, q2post, qdpost,
                   py, pvalid);
          if (hiHalf) {
#pragma unroll
            for (int j = 0; j < 8; ++j) pp[8 + j][qq][16 * hf + ss] = (bf16)py[j];
          } else {
#pragma unroll
            for (int j = 0; j < 8; ++j) pp[j][qq][16 * hf + ss] = (bf16)py[j];
          }
        }
      } else {
        for (int i = tid; i < 4096; i += 512) {
          const int nn = i >> 8, qr = (i >> 4) & 15, sv = i & 15;
          pp[nn][qr][16 * hf + sv] = (bf16)0.f;
        }
      }
      __syncthreads();
    }

    // P(16x32) * V(32x16 per feature tile)
    v16bf pa;
    {
      const v8bf* pr8 = (const v8bf*)(&pp[n][r][0]);   // row = 32 bf16, 64B
      v8bf lo = pr8[2 * h];      // elems 0..7  -> keys 16h+e
      v8bf hi = pr8[2 * h + 1];  // elems 8..15 -> keys 16h+8+e
#pragma unroll
      for (int e = 0; e < 8; ++e) { pa[e] = lo[e]; pa[8 + e] = hi[e]; }
    }
    if (s0p >= 0 && s0p + 31 < T_SEQ) {
      // fast path: all 32 key rows in range -> immediate-offset b32 loads
      const float* vb0 = v + (((size_t)(b * T_SEQ + s0p + 16 * h)) * NHEADS + n) * DHEAD + r;
#pragma unroll
      for (int ft = 0; ft < 8; ++ft) {
        v16bf vbf;
#pragma unroll
        for (int e = 0; e < 16; ++e)
          vbf[e] = (bf16)vb0[(size_t)e * (NHEADS * DHEAD) + 16 * ft];
        oacc[ft] = __builtin_amdgcn_wmma_f32_16x16x32_bf16(false, pa, false, vbf,
                                                           (short)0, oacc[ft], false, false);
      }
    } else {
      // edge path: clamp rows; clamped rows pair with zero probabilities
#pragma unroll
      for (int ft = 0; ft < 8; ++ft) {
        v16bf vbf;
#pragma unroll
        for (int e = 0; e < 16; ++e) {
          int key = s0p + 16 * h + e;
          key = key < 0 ? 0 : (key > T_SEQ - 1 ? T_SEQ - 1 : key);
          vbf[e] = (bf16)v[(((size_t)(b * T_SEQ + key)) * NHEADS + n) * DHEAD + 16 * ft + r];
        }
        oacc[ft] = __builtin_amdgcn_wmma_f32_16x16x32_bf16(false, pa, false, vbf,
                                                           (short)0, oacc[ft], false, false);
      }
    }
  }

  // ---- write out[b, t, n, d]; C/D layout: vgpr rr -> row rr+8h, lane -> col
#pragma unroll
  for (int ft = 0; ft < 8; ++ft)
#pragma unroll
    for (int rr = 0; rr < 8; ++rr)
      out[(((size_t)(b * T_SEQ + t0 + rr + 8 * h)) * NHEADS + n) * DHEAD + 16 * ft + r] =
          oacc[ft][rr];
}

extern "C" void kernel_launch(void* const* d_in, const int* in_sizes, int n_in,
                              void* d_out, int out_size, void* d_ws, size_t ws_size,
                              hipStream_t stream) {
  (void)in_sizes; (void)n_in; (void)d_ws; (void)ws_size; (void)out_size;
  const float* q        = (const float*)d_in[0];
  const float* k        = (const float*)d_in[1];
  const float* v        = (const float*)d_in[2];
  const float* w_pre    = (const float*)d_in[3];
  const float* w_post   = (const float*)d_in[4];
  const float* qw1_pre  = (const float*)d_in[5];
  const float* qw2_pre  = (const float*)d_in[6];
  const float* kw1_pre  = (const float*)d_in[7];
  const float* kw2_pre  = (const float*)d_in[8];
  const float* qw1_post = (const float*)d_in[9];
  const float* qw2_post = (const float*)d_in[10];
  const float* kw1_post = (const float*)d_in[11];
  const float* kw2_post = (const float*)d_in[12];
  const float* qdd_pre  = (const float*)d_in[13];
  const float* kdd_pre  = (const float*)d_in[14];
  const float* qdd_post = (const float*)d_in[15];
  const float* kdd_post = (const float*)d_in[16];
  float* out = (float*)d_out;

  dim3 grid(2 * (T_SEQ / 16));   // B * T/16 = 128 workgroups
  dim3 block(512);               // 16 waves, one head per wave
  hipLaunchKernelGGL(slide_talking_heads_attn, grid, block, 0, stream,
                     q, k, v, w_pre, w_post,
                     qw1_pre, qw2_pre, kw1_pre, kw2_pre,
                     qw1_post, qw2_post, kw1_post, kw2_post,
                     qdd_pre, kdd_pre, qdd_post, kdd_post, out);
}
